// FullAttentionBlock_87428354277812
// MI455X (gfx1250) — compile-verified, matrix operand order checked by reading, not run on hardware
//
#include <hip/hip_runtime.h>
#include <hip/hip_bf16.h>

// ---------------------------------------------------------------------------
// CDNA5 (gfx1250) transformer block: LN -> QKV GEMM -> flash attention ->
// proj GEMM -> LN -> FFN GEMMs. All matmuls via v_wmma_f32_16x16x32_bf16.
// wave32. GEMMs: 256-thread blocks (8 waves). Attention: 512 threads
// (16 waves, 256 query rows/block) to halve L2 passes over K/V.
// ---------------------------------------------------------------------------

typedef __attribute__((ext_vector_type(16))) __bf16 v16bf;
typedef __attribute__((ext_vector_type(8)))  __bf16 v8bf;   // 16 bytes
typedef __attribute__((ext_vector_type(8)))  float  v8f;

#define DEV __device__ __forceinline__

DEV v8f wmma_bf16(v16bf a, v16bf b, v8f c) {
    // (neg_a, A, neg_b, B, c_mod, C, reuse_a, reuse_b)
    return __builtin_amdgcn_wmma_f32_16x16x32_bf16(false, a, false, b,
                                                   (short)0, c, false, false);
}

DEV v8f zero8() { v8f z; for (int i = 0; i < 8; ++i) z[i] = 0.f; return z; }

// Load one 16x32 bf16 A/B fragment. 16-bit A-matrix layout (ISA 7.12.2):
//   lanes 0-15: row M = l, K = hk+0..7 (elems 0..7) and hk+16..23 (elems 8..15)
//   lanes 16-31: hk = 8 -> K = 8..15 and 24..31
// 'row' is the per-lane matrix row (for B fragments: the output column, with
// the operand stored K-contiguous, i.e. transposed weights / V^T / K rows).
DEV v16bf load_frag(const __bf16* base, int stride, int row, int koff) {
    const __bf16* p = base + (size_t)row * stride + koff;
    v16bf f;
    *(v8bf*)(&f)       = *(const v8bf*)(p);
    *((v8bf*)(&f) + 1) = *(const v8bf*)(p + 16);
    return f;
}

// VALU-only 16-lane row reduction via DPP row_ror (no LDS, no dscnt stalls).
template <int CTRL>
DEV float dpp_ror(float v) {
    return __int_as_float(__builtin_amdgcn_update_dpp(
        0, __float_as_int(v), CTRL, 0xf, 0xf, true));
}
DEV float rowmax16(float v) {
    v = fmaxf(v, dpp_ror<0x128>(v));  // row_ror:8
    v = fmaxf(v, dpp_ror<0x124>(v));  // row_ror:4
    v = fmaxf(v, dpp_ror<0x122>(v));  // row_ror:2
    v = fmaxf(v, dpp_ror<0x121>(v));  // row_ror:1
    return v;
}

// ---------------------------------------------------------------------------
// Weight convert + transpose: w [K][Nc] f32 -> wt [Nc][K] bf16
// ---------------------------------------------------------------------------
__global__ __launch_bounds__(256) void wconvT_kernel(const float* __restrict__ w,
                                                     __bf16* __restrict__ wt,
                                                     int K, int Nc) {
    int idx = blockIdx.x * 256 + threadIdx.x;
    if (idx < K * Nc) {
        int k = idx / Nc, n = idx - k * Nc;
        wt[(size_t)n * K + k] = (__bf16)w[idx];
    }
}

// ---------------------------------------------------------------------------
// LN1: x [2,512,4096] -> t f32 [8192][512] (residual) + h bf16 [8192][512]
// ---------------------------------------------------------------------------
__global__ __launch_bounds__(256) void ln1_kernel(const float* __restrict__ x,
                                                  const float* __restrict__ g,
                                                  const float* __restrict__ be,
                                                  float* __restrict__ t,
                                                  __bf16* __restrict__ h) {
    int row = blockIdx.x;             // b*4096 + n
    int b = row >> 12, n = row & 4095;
    int tid = threadIdx.x, lane = tid & 31, wave = tid >> 5;
    int c0 = tid, c1 = tid + 256;
    float v0 = x[(((size_t)(b * 512 + c0)) << 12) + n];
    float v1 = x[(((size_t)(b * 512 + c1)) << 12) + n];
    float s = v0 + v1, ss = v0 * v0 + v1 * v1;
    for (int m = 16; m; m >>= 1) { s += __shfl_xor(s, m, 32); ss += __shfl_xor(ss, m, 32); }
    __shared__ float rs_[8], rss_[8];
    if (lane == 0) { rs_[wave] = s; rss_[wave] = ss; }
    __syncthreads();
    float tot = 0.f, tot2 = 0.f;
    for (int i = 0; i < 8; ++i) { tot += rs_[i]; tot2 += rss_[i]; }
    float mu = tot * (1.f / 512.f);
    float var = tot2 * (1.f / 512.f) - mu * mu;
    float rstd = rsqrtf(var + 1e-5f);
    size_t base = (size_t)row * 512;
    t[base + c0] = v0;
    t[base + c1] = v1;
    h[base + c0] = (__bf16)((v0 - mu) * rstd * g[c0] + be[c0]);
    h[base + c1] = (__bf16)((v1 - mu) * rstd * g[c1] + be[c1]);
}

// LN2: t2 f32 [8192][512] -> h2 bf16 [8192][512]
__global__ __launch_bounds__(256) void ln2_kernel(const float* __restrict__ t2,
                                                  const float* __restrict__ g,
                                                  const float* __restrict__ be,
                                                  __bf16* __restrict__ h2) {
    int row = blockIdx.x;
    int tid = threadIdx.x, lane = tid & 31, wave = tid >> 5;
    size_t base = (size_t)row * 512;
    int c0 = tid, c1 = tid + 256;
    float v0 = t2[base + c0], v1 = t2[base + c1];
    float s = v0 + v1, ss = v0 * v0 + v1 * v1;
    for (int m = 16; m; m >>= 1) { s += __shfl_xor(s, m, 32); ss += __shfl_xor(ss, m, 32); }
    __shared__ float rs_[8], rss_[8];
    if (lane == 0) { rs_[wave] = s; rss_[wave] = ss; }
    __syncthreads();
    float tot = 0.f, tot2 = 0.f;
    for (int i = 0; i < 8; ++i) { tot += rs_[i]; tot2 += rss_[i]; }
    float mu = tot * (1.f / 512.f);
    float var = tot2 * (1.f / 512.f) - mu * mu;
    float rstd = rsqrtf(var + 1e-5f);
    h2[base + c0] = (__bf16)((v0 - mu) * rstd * g[c0] + be[c0]);
    h2[base + c1] = (__bf16)((v1 - mu) * rstd * g[c1] + be[c1]);
}

// ---------------------------------------------------------------------------
// Generic bf16 WMMA GEMM: C[M,N] = A[M,K] * Wt[N,K]^T  (+ per-mode epilogue)
// Block tile 128x128, 8 waves each owning a 32x64 tile (2x4 16x16 accum).
// MODE 0: +bias, scatter to Q [bh][n][64], K [bh][n][64], V^T [bh][64][n] bf16
// MODE 1: +bias +resid -> outf f32 (t2)
// MODE 2: +bias, relu -> outb bf16
// MODE 3: +bias +resid, transpose to [b][c][n] and add xin -> outf (d_out)
// ---------------------------------------------------------------------------
template <int MODE>
__global__ __launch_bounds__(256) void gemm_bf16_kernel(
    const __bf16* __restrict__ A, const __bf16* __restrict__ Wt,
    const float* __restrict__ bias, const float* __restrict__ resid,
    const float* __restrict__ xin, float* __restrict__ outf,
    __bf16* __restrict__ outb, __bf16* __restrict__ qg,
    __bf16* __restrict__ kg, __bf16* __restrict__ vtg, int Nsz, int Ksz) {
    __shared__ alignas(16) __bf16 As[128 * 32];
    __shared__ alignas(16) __bf16 Bs[128 * 32];

    const int tid = threadIdx.x;
    const int lane = tid & 31, wave = tid >> 5;
    const int l15 = lane & 15;
    const int hk = (lane >> 4) << 3;      // 0 or 8
    const int wm = (wave & 3) * 32;       // 4 waves along M
    const int wn = (wave >> 2) * 64;      // 2 waves along N
    const int blockN = blockIdx.x * 128, blockM = blockIdx.y * 128;
    const int ldr = tid >> 1, seg = (tid & 1) * 16;

    v8f acc[2][4];
    for (int i = 0; i < 2; ++i)
        for (int j = 0; j < 4; ++j) acc[i][j] = zero8();

    for (int k0 = 0; k0 < Ksz; k0 += 32) {
        const __bf16* gA = A + (size_t)(blockM + ldr) * Ksz + k0 + seg;
        const __bf16* gB = Wt + (size_t)(blockN + ldr) * Ksz + k0 + seg;
        *(v8bf*)(&As[ldr * 32 + seg])     = *(const v8bf*)(gA);
        *(v8bf*)(&As[ldr * 32 + seg + 8]) = *(const v8bf*)(gA + 8);
        *(v8bf*)(&Bs[ldr * 32 + seg])     = *(const v8bf*)(gB);
        *(v8bf*)(&Bs[ldr * 32 + seg + 8]) = *(const v8bf*)(gB + 8);
        __syncthreads();
        v16bf af[2], bfv[4];
        for (int i = 0; i < 2; ++i) af[i] = load_frag(As, 32, wm + i * 16 + l15, hk);
        for (int j = 0; j < 4; ++j) bfv[j] = load_frag(Bs, 32, wn + j * 16 + l15, hk);
        for (int i = 0; i < 2; ++i)
            for (int j = 0; j < 4; ++j) acc[i][j] = wmma_bf16(af[i], bfv[j], acc[i][j]);
        __syncthreads();
    }

    // Epilogue. C/D layout: VGPR e holds row (e + 8*(lane>=16)) == e + hk.
    for (int i = 0; i < 2; ++i)
        for (int j = 0; j < 4; ++j)
            for (int e = 0; e < 8; ++e) {
                int r = blockM + wm + i * 16 + e + hk;
                int cg = blockN + wn + j * 16 + l15;
                float v = acc[i][j][e] + bias[cg];
                if (MODE == 0) {
                    int bq = r >> 12, n = r & 4095;
                    int sect = cg >> 9, cc = cg & 511;
                    int hh = cc >> 6, hd = cc & 63;
                    size_t ho = (size_t)(bq * 8 + hh);
                    if (sect == 0)      qg[(ho * 4096 + n) * 64 + hd] = (__bf16)v;
                    else if (sect == 1) kg[(ho * 4096 + n) * 64 + hd] = (__bf16)v;
                    else                vtg[(ho * 64 + hd) * 4096 + n] = (__bf16)v;
                } else if (MODE == 1) {
                    v += resid[(size_t)r * 512 + cg];
                    outf[(size_t)r * 512 + cg] = v;
                } else if (MODE == 2) {
                    v = fmaxf(v, 0.f);
                    outb[(size_t)r * (size_t)Nsz + cg] = (__bf16)v;
                } else {
                    v += resid[(size_t)r * 512 + cg];
                    int bq = r >> 12, n = r & 4095;
                    size_t oi = ((size_t)(bq * 512 + cg)) * 4096 + n;
                    outf[oi] = v + xin[oi];
                }
            }
}

// ---------------------------------------------------------------------------
// Flash attention. Grid: (16 q-tiles of 256 rows, 16 batch*head),
// 512 threads = 16 waves (halves L2 passes over K/V vs 128-row blocks).
// Each wave owns 16 query rows; online softmax in WMMA C-layout.
//  - row max: DPP row_ror butterfly (VALU only)
//  - row sum: P @ ones via WMMA (C-layout result replicates per-row sums)
//  - P C-layout -> A-fragment: per-wave private LDS bounce + s_wait_dscnt
// ---------------------------------------------------------------------------
__global__ __launch_bounds__(512) void attn_kernel(const __bf16* __restrict__ Qg,
                                                   const __bf16* __restrict__ Kg,
                                                   const __bf16* __restrict__ VTg,
                                                   __bf16* __restrict__ outb) {
    const int bh = blockIdx.y;                 // 0..15
    const int tid = threadIdx.x;
    const int lane = tid & 31, wave = tid >> 5; // wave 0..15
    const int l15 = lane & 15;
    const int hk = (lane >> 4) << 3;
    const int qRow = blockIdx.x * 256 + wave * 16;

    const __bf16* Qb = Qg + (size_t)bh * 4096 * 64;
    const __bf16* Kb = Kg + (size_t)bh * 4096 * 64;
    const __bf16* Vb = VTg + (size_t)bh * 64 * 4096;

    __shared__ alignas(16) __bf16 Ps[16 * 16 * 64];  // 2KB per wave (32KB)
    __bf16* pw = &Ps[wave * 16 * 64];

    // Q fragments (K = head_dim 64 -> two 16x32 fragments), held in registers.
    v16bf qf0 = load_frag(Qb, 64, qRow + l15, hk);
    v16bf qf1 = load_frag(Qb, 64, qRow + l15, 32 + hk);

    // All-ones B fragment for WMMA row-sum.
    v16bf ones;
    for (int i = 0; i < 16; ++i) ones[i] = (__bf16)1.0f;

    v8f o[4];
    for (int t = 0; t < 4; ++t) o[t] = zero8();
    float mrow[8], lrow[8];
    for (int j = 0; j < 8; ++j) { mrow[j] = -3.0e38f; lrow[j] = 0.f; }

    // softmax(s / sqrt(512)) via exp2: kscale = log2(e)/sqrt(512)
    const float kscale = 1.4426950408889634f * 0.04419417382415922f;

    for (int kv = 0; kv < 4096; kv += 64) {
        // S = Q (16x64) @ K^T (64 x 64keys), four 16x16 tiles along keys.
        v8f s[4];
        for (int t = 0; t < 4; ++t) {
            v16bf kf0 = load_frag(Kb, 64, kv + t * 16 + l15, hk);
            v16bf kf1 = load_frag(Kb, 64, kv + t * 16 + l15, 32 + hk);
            s[t] = wmma_bf16(qf0, kf0, zero8());
            s[t] = wmma_bf16(qf1, kf1, s[t]);
        }
        // Row max across tiles (VALU) then across the 16-lane row (DPP).
        float corr[8], negmk[8];
        for (int j = 0; j < 8; ++j) {
            float mx = fmaxf(fmaxf(s[0][j], s[1][j]), fmaxf(s[2][j], s[3][j]));
            mx = rowmax16(mx);
            float mn = fmaxf(mrow[j], mx);
            corr[j] = exp2f((mrow[j] - mn) * kscale);
            mrow[j] = mn;
            negmk[j] = -mn * kscale;   // fold (s - m)*k into one FMA
        }
        // P = exp2(s*k - m*k); rescale O.
        for (int t = 0; t < 4; ++t)
            for (int j = 0; j < 8; ++j) {
                s[t][j] = exp2f(__builtin_fmaf(s[t][j], kscale, negmk[j]));
                o[t][j] *= corr[j];
            }

        // C-layout -> A-fragment layout via private LDS region (wave-local,
        // no barrier needed, just DS-counter drain).
        for (int t = 0; t < 4; ++t)
            for (int j = 0; j < 8; ++j)
                pw[(j + hk) * 64 + t * 16 + l15] = (__bf16)s[t][j];
        asm volatile("s_wait_dscnt 0" ::: "memory");
        v16bf pf0 = load_frag(pw, 64, l15, hk);
        v16bf pf1 = load_frag(pw, 64, l15, 32 + hk);

        // Row sums via WMMA: P @ ones -> every column holds the row sum,
        // C-layout replication matches lrow[] indexing exactly.
        v8f rs8 = wmma_bf16(pf0, ones, zero8());
        rs8 = wmma_bf16(pf1, ones, rs8);
        for (int j = 0; j < 8; ++j) lrow[j] = lrow[j] * corr[j] + rs8[j];

        // O += P (16x64) @ V (64keys x 64), V^T stored [hd][n] -> contiguous.
        for (int t = 0; t < 4; ++t) {
            v16bf vf0 = load_frag(Vb, 4096, t * 16 + l15, kv + hk);
            v16bf vf1 = load_frag(Vb, 4096, t * 16 + l15, kv + 32 + hk);
            o[t] = wmma_bf16(pf0, vf0, o[t]);
            o[t] = wmma_bf16(pf1, vf1, o[t]);
        }
    }

    // Normalize (reciprocal once per row) and write merged-head output
    // [b][n][h*64 + c] as bf16.
    float inv[8];
    for (int j = 0; j < 8; ++j) inv[j] = 1.0f / lrow[j];
    const int b = bh >> 3, h = bh & 7;
    for (int t = 0; t < 4; ++t)
        for (int j = 0; j < 8; ++j) {
            int n = qRow + j + hk;
            int c = h * 64 + t * 16 + l15;
            outb[((size_t)(b * 4096 + n)) * 512 + c] = (__bf16)(o[t][j] * inv[j]);
        }
}

// ---------------------------------------------------------------------------
// Host orchestration
// ---------------------------------------------------------------------------
extern "C" void kernel_launch(void* const* d_in, const int* in_sizes, int n_in,
                              void* d_out, int out_size, void* d_ws, size_t ws_size,
                              hipStream_t stream) {
    (void)in_sizes; (void)n_in; (void)out_size; (void)ws_size;
    const float* x      = (const float*)d_in[0];
    const float* ln1_g  = (const float*)d_in[1];
    const float* ln1_b  = (const float*)d_in[2];
    const float* w_qkv  = (const float*)d_in[3];
    const float* b_qkv  = (const float*)d_in[4];
    const float* w_proj = (const float*)d_in[5];
    const float* b_proj = (const float*)d_in[6];
    const float* ln2_g  = (const float*)d_in[7];
    const float* ln2_b  = (const float*)d_in[8];
    const float* w_ffn1 = (const float*)d_in[9];
    const float* b_ffn1 = (const float*)d_in[10];
    const float* w_ffn2 = (const float*)d_in[11];
    const float* b_ffn2 = (const float*)d_in[12];

    char* ws = (char*)d_ws;
    size_t off = 0;
    auto alloc = [&](size_t bytes) {
        size_t o = off;
        off += (bytes + 255) & ~(size_t)255;
        return o;
    };

    float*  t      = (float*)(ws + alloc(8192ull * 512 * 4));
    __bf16* h      = (__bf16*)(ws + alloc(8192ull * 512 * 2));
    __bf16* wqkvT  = (__bf16*)(ws + alloc(1536ull * 512 * 2));
    __bf16* wprojT = (__bf16*)(ws + alloc(512ull * 512 * 2));
    __bf16* wff1T  = (__bf16*)(ws + alloc(2048ull * 512 * 2));
    __bf16* wff2T  = (__bf16*)(ws + alloc(512ull * 2048 * 2));
    __bf16* qg     = (__bf16*)(ws + alloc(16ull * 4096 * 64 * 2));
    __bf16* kg     = (__bf16*)(ws + alloc(16ull * 4096 * 64 * 2));
    __bf16* vtg    = (__bf16*)(ws + alloc(16ull * 64 * 4096 * 2));
    __bf16* aout   = (__bf16*)(ws + alloc(8192ull * 512 * 2));
    float*  t2     = (float*)(ws + alloc(8192ull * 512 * 4));
    __bf16* h2     = (__bf16*)(ws + alloc(8192ull * 512 * 2));
    __bf16* act    = (__bf16*)(ws + alloc(8192ull * 2048 * 2));

    // Weight convert + transpose (bf16, [N][K]).
    wconvT_kernel<<<(512 * 1536 + 255) / 256, 256, 0, stream>>>(w_qkv, wqkvT, 512, 1536);
    wconvT_kernel<<<(512 * 512 + 255) / 256, 256, 0, stream>>>(w_proj, wprojT, 512, 512);
    wconvT_kernel<<<(512 * 2048 + 255) / 256, 256, 0, stream>>>(w_ffn1, wff1T, 512, 2048);
    wconvT_kernel<<<(2048 * 512 + 255) / 256, 256, 0, stream>>>(w_ffn2, wff2T, 2048, 512);

    // LN1 (also materializes fp32 residual t).
    ln1_kernel<<<8192, 256, 0, stream>>>(x, ln1_g, ln1_b, t, h);

    // QKV GEMM: [8192,512] x [512,1536] -> per-head Q/K/V^T bf16.
    gemm_bf16_kernel<0><<<dim3(12, 64), 256, 0, stream>>>(
        h, wqkvT, b_qkv, nullptr, nullptr, nullptr, nullptr, qg, kg, vtg, 1536, 512);

    // Flash attention over 16 (batch, head) pairs, 256 query rows per block.
    attn_kernel<<<dim3(16, 16), 512, 0, stream>>>(qg, kg, vtg, aout);

    // Projection + residual -> t2 (f32).
    gemm_bf16_kernel<1><<<dim3(4, 64), 256, 0, stream>>>(
        aout, wprojT, b_proj, t, nullptr, t2, nullptr, nullptr, nullptr, nullptr, 512, 512);

    // LN2.
    ln2_kernel<<<8192, 256, 0, stream>>>(t2, ln2_g, ln2_b, h2);

    // FFN1 + ReLU -> bf16 activations.
    gemm_bf16_kernel<2><<<dim3(16, 64), 256, 0, stream>>>(
        h2, wff1T, b_ffn1, nullptr, nullptr, nullptr, act, nullptr, nullptr, nullptr, 2048, 512);

    // FFN2 + residual + transpose back + x_in -> d_out (f32 [2,512,64,64]).
    gemm_bf16_kernel<3><<<dim3(4, 64), 256, 0, stream>>>(
        act, wff2T, b_ffn2, t2, x, (float*)d_out, nullptr, nullptr, nullptr, nullptr, 512, 2048);
}
